// HGTPolicy_53051436040798
// MI455X (gfx1250) — matrix operand order repaired and checked by reading, not compiled, not run on hardware
//
#include <hip/hip_runtime.h>

// ---------------------------------------------------------------------------
// HGT policy network forward for MI455X (gfx1250, wave32).
// Dense GEMMs: v_wmma_f32_16x16x32_f16 with f16 operands pre-staged in
// fragment-friendly layouts (A row-major f16, B transposed to [N][K] f16)
// so each wave builds fragments from contiguous global_load_b128s.
// Each wave computes a 16x32 output tile (2 WMMAs per K-step share the A
// fragment -> halved A traffic, doubled MAC density per loop body).
// Edge phase: 3-pass segment softmax with float atomics (L2-resident).
// Workspace usage ~140 MB (fits the 192 MB L2 of MI455X).
// ---------------------------------------------------------------------------

typedef __attribute__((ext_vector_type(16))) _Float16 v16h;
typedef __attribute__((ext_vector_type(8)))  _Float16 v8h;
typedef __attribute__((ext_vector_type(4)))  _Float16 v4h;
typedef __attribute__((ext_vector_type(8)))  float    v8f;

#define HDIM 64
#define NHEAD 8
#define DHEAD 8

__device__ __forceinline__ float geluf(float x) {
    return 0.5f * x * (1.0f + erff(x * 0.70710678118654752f));
}

// float atomic max via integer ordering trick (amax initialized to -3e38)
__device__ __forceinline__ void atomicMaxF(float* addr, float v) {
    if (v >= 0.0f) atomicMax((int*)addr, __float_as_int(v));
    else           atomicMin((unsigned int*)addr, (unsigned int)__float_as_int(v));
}

// ---------------------------------------------------------------------------
__global__ void k_fill(float* __restrict__ p, long n, float v) {
    long i = (long)blockIdx.x * blockDim.x + threadIdx.x;
    if (i < n) p[i] = v;
}

// f32 -> f16 streaming conversion (optionally fused exact GELU), 4-wide
__global__ void k_cvt16(const float* __restrict__ s, _Float16* __restrict__ d,
                        long n, int dogelu) {
    long i = ((long)blockIdx.x * blockDim.x + threadIdx.x) * 4;
    if (i + 3 < n) {
        float4 v = *(const float4*)(s + i);
        if (dogelu) { v.x = geluf(v.x); v.y = geluf(v.y); v.z = geluf(v.z); v.w = geluf(v.w); }
        v4h o;
        o[0] = (_Float16)v.x; o[1] = (_Float16)v.y;
        o[2] = (_Float16)v.z; o[3] = (_Float16)v.w;
        *(v4h*)(d + i) = o;
    } else {
        for (long j = i; j < n; ++j) {
            float x = s[j];
            if (dogelu) x = geluf(x);
            d[j] = (_Float16)x;
        }
    }
}

// weight prep: W[K,N] f32 row-major -> Wt[N,K] f16 (transposed, contiguous K)
__global__ void k_wt16(const float* __restrict__ W, _Float16* __restrict__ Wt,
                       int K, int N) {
    long t = (long)blockIdx.x * blockDim.x + threadIdx.x;
    if (t >= (long)N * K) return;
    int n = (int)(t / K);
    int k = (int)(t % K);
    Wt[t] = (_Float16)W[(size_t)k * N + n];
}

// input projection y = x@W + b (F<=8 features) + per-type sum/sumsq stats
__global__ void k_proj_stats(const float* __restrict__ x, const float* __restrict__ W,
                             const float* __restrict__ b, int N, int F,
                             float* __restrict__ X, float* __restrict__ stats) {
    __shared__ float s1[256];
    __shared__ float s2[256];
    long t = (long)blockIdx.x * blockDim.x + threadIdx.x;
    float y = 0.0f;
    bool valid = (t < (long)N * HDIM);
    if (valid) {
        int n = (int)(t >> 6);
        int j = (int)(t & 63);
        y = b[j];
        for (int f = 0; f < F; ++f) y += x[(size_t)n * F + f] * W[f * HDIM + j];
        X[(size_t)n * HDIM + j] = y;
    }
    s1[threadIdx.x] = valid ? y : 0.0f;
    s2[threadIdx.x] = valid ? y * y : 0.0f;
    __syncthreads();
    for (int st = 128; st > 0; st >>= 1) {
        if ((int)threadIdx.x < st) {
            s1[threadIdx.x] += s1[threadIdx.x + st];
            s2[threadIdx.x] += s2[threadIdx.x + st];
        }
        __syncthreads();
    }
    if (threadIdx.x == 0) {
        atomicAdd(&stats[0], s1[0]);
        atomicAdd(&stats[1], s2[0]);
    }
}

// graph-norm: normalize over all N*H elements of this type
__global__ void k_norm(float* __restrict__ X, int N, const float* __restrict__ stats,
                       const float* __restrict__ gamma, const float* __restrict__ beta) {
    long t = (long)blockIdx.x * blockDim.x + threadIdx.x;
    if (t >= (long)N * HDIM) return;
    float cnt = (float)N * (float)HDIM;
    float mu  = stats[0] / cnt;
    float var = stats[1] / cnt - mu * mu;
    float sd  = sqrtf(fmaxf(var, 0.0f));
    int j = (int)(t & 63);
    X[t] = (X[t] - mu) / (sd + 1e-5f) * gamma[j] + beta[j];
}

// ---------------------------------------------------------------------------
// Wave-level WMMA GEMM: C[M,N] = act(A[M,K] @ B[K,N] + bias).
// A: f16 row-major [M,K] (activation pre-applied in the cvt pass).
// Bt: f16 [N,K] (B transposed) -> every fragment is contiguous b128 loads.
// Each wave owns a 16x32 tile: one A fragment feeds two WMMAs per K-step.
// N must be a multiple of 32 (true for 192/64/32 here).
// MODE: 0=none, 1=relu, 2=sigmoid(skip)-gated residual with Xold (f32).
// ---------------------------------------------------------------------------
template <int MODE>
__global__ void k_gemm_wmma(const _Float16* __restrict__ A, const _Float16* __restrict__ Bt,
                            const float* __restrict__ bias, float* __restrict__ C,
                            int M, int N, int K,
                            const float* __restrict__ Xold,
                            const float* __restrict__ skipPtr) {
    const int lane = threadIdx.x & 31;
    const int mtiles = (M + 15) >> 4;
    const int ntiles = N >> 5;                     // 32-wide column tiles
    int tile = blockIdx.x * (blockDim.x >> 5) + (threadIdx.x >> 5);
    if (tile >= mtiles * ntiles) return;           // uniform per wave: EXEC stays all-ones
    const int mt = tile / ntiles;
    const int nt = tile - mt * ntiles;

    const int r  = lane & 15;                      // row (A) / col (B,C) within tile
    const int hs = lane >> 4;                      // half-wave selector

    int arow = mt * 16 + r;
    if (arow > M - 1) arow = M - 1;                // clamp: duplicate rows, stores guarded
    const _Float16* Ar  = A  + (size_t)arow * K;
    const _Float16* Bc0 = Bt + (size_t)(nt * 32 + r) * K;        // B columns [nt*32 .. +16)
    const _Float16* Bc1 = Bt + (size_t)(nt * 32 + 16 + r) * K;   // B columns [nt*32+16 .. +32)

    v8f acc0 = {0.f, 0.f, 0.f, 0.f, 0.f, 0.f, 0.f, 0.f};
    v8f acc1 = {0.f, 0.f, 0.f, 0.f, 0.f, 0.f, 0.f, 0.f};
    for (int k0 = 0; k0 < K; k0 += 32) {
        if (k0 + 32 < K) {                         // near-cache prefetch of next K-step
            __builtin_prefetch(Ar  + k0 + 32, 0, 3);   // global_prefetch_b8
            __builtin_prefetch(Bc0 + k0 + 32, 0, 3);
            __builtin_prefetch(Bc1 + k0 + 32, 0, 3);
        }
        // A fragment (16x32 f16): contiguous 32B per lane (k = 16*hs + s)
        v8h a0 = *(const v8h*)(Ar + k0 + hs * 16);
        v8h a1 = *(const v8h*)(Ar + k0 + hs * 16 + 8);
        // B fragments (32x16 f16 each): contiguous 32B per lane
        v8h b00 = *(const v8h*)(Bc0 + k0 + hs * 16);
        v8h b01 = *(const v8h*)(Bc0 + k0 + hs * 16 + 8);
        v8h b10 = *(const v8h*)(Bc1 + k0 + hs * 16);
        v8h b11 = *(const v8h*)(Bc1 + k0 + hs * 16 + 8);
        v16h af, bf0, bf1;
        #pragma unroll
        for (int i = 0; i < 8; ++i) {
            af[i] = a0[i];   af[i + 8]  = a1[i];
            bf0[i] = b00[i]; bf0[i + 8] = b01[i];
            bf1[i] = b10[i]; bf1[i + 8] = b11[i];
        }
        acc0 = __builtin_amdgcn_wmma_f32_16x16x32_f16(false, af, false, bf0,
                                                      (short)0, acc0, false, false);
        acc1 = __builtin_amdgcn_wmma_f32_16x16x32_f16(false, af, false, bf1,
                                                      (short)0, acc1, false, false);
    }

    const int col0 = nt * 32 + r;
    const int col1 = col0 + 16;
    const float bv0 = bias ? bias[col0] : 0.0f;
    const float bv1 = bias ? bias[col1] : 0.0f;
    float gate = 0.0f;
    if (MODE == 2) gate = 1.0f / (1.0f + expf(-skipPtr[0]));
    // C fragment: VGPR j -> row j (lanes 0-15) / row j+8 (lanes 16-31)
    #pragma unroll
    for (int j = 0; j < 8; ++j) {
        int row = mt * 16 + hs * 8 + j;
        if (row < M) {
            float v0 = acc0[j] + bv0;
            float v1 = acc1[j] + bv1;
            if (MODE == 1) { v0 = fmaxf(v0, 0.0f); v1 = fmaxf(v1, 0.0f); }
            if (MODE == 2) {
                v0 = gate * v0 + (1.0f - gate) * Xold[(size_t)row * N + col0];
                v1 = gate * v1 + (1.0f - gate) * Xold[(size_t)row * N + col1];
            }
            C[(size_t)row * N + col0] = v0;
            C[(size_t)row * N + col1] = v1;
        }
    }
}

// ---------------------------------------------------------------------------
// Edge-phase kernels: one thread per (edge, head).
// KQV rows: [k(64) | q(64) | v(64)], stride 192.
// ---------------------------------------------------------------------------
__global__ void k_edge_alpha(const int* __restrict__ ed, int E, int srcOff, int dstOff,
                             const float* __restrict__ KQV,
                             const float* __restrict__ krel,   // [h][d][e'] 512 floats
                             const float* __restrict__ prel,   // [h] 8 floats
                             float* __restrict__ alpha, float* __restrict__ amax) {
    long t = (long)blockIdx.x * blockDim.x + threadIdx.x;
    if (t >= (long)E * NHEAD) return;
    int e = (int)(t >> 3), h = (int)(t & 7);
    int sN = srcOff + ed[e];
    int dN = dstOff + ed[E + e];
    const float* kp = KQV + (size_t)sN * 192 + h * DHEAD;
    const float* qp = KQV + (size_t)dN * 192 + 64 + h * DHEAD;
    const float* kr = krel + h * 64;
    float kv[8];
    #pragma unroll
    for (int i = 0; i < 8; ++i) kv[i] = kp[i];
    float a = 0.0f;
    #pragma unroll
    for (int eo = 0; eo < 8; ++eo) {
        float ke = 0.0f;
        #pragma unroll
        for (int di = 0; di < 8; ++di) ke += kv[di] * kr[di * 8 + eo];
        a += qp[eo] * ke;
    }
    a *= prel[h] * 0.35355339059327373f;           // 1/sqrt(D=8)
    alpha[(size_t)e * NHEAD + h] = a;
    atomicMaxF(amax + (size_t)dN * NHEAD + h, a);
}

__global__ void k_edge_den(const int* __restrict__ ed, int E, int dstOff,
                           const float* __restrict__ alpha, const float* __restrict__ amax,
                           float* __restrict__ den) {
    long t = (long)blockIdx.x * blockDim.x + threadIdx.x;
    if (t >= (long)E * NHEAD) return;
    int e = (int)(t >> 3), h = (int)(t & 7);
    int dN = dstOff + ed[E + e];
    float ex = expf(alpha[(size_t)e * NHEAD + h] - amax[(size_t)dN * NHEAD + h]);
    atomicAdd(den + (size_t)dN * NHEAD + h, ex);
}

__global__ void k_edge_agg(const int* __restrict__ ed, int E, int srcOff, int dstOff,
                           const float* __restrict__ KQV, const float* __restrict__ vrel,
                           const float* __restrict__ alpha, const float* __restrict__ amax,
                           const float* __restrict__ den, float* __restrict__ agg) {
    long t = (long)blockIdx.x * blockDim.x + threadIdx.x;
    if (t >= (long)E * NHEAD) return;
    int e = (int)(t >> 3), h = (int)(t & 7);
    int sN = srcOff + ed[e];
    int dN = dstOff + ed[E + e];
    float ex = expf(alpha[(size_t)e * NHEAD + h] - amax[(size_t)dN * NHEAD + h]);
    float w  = ex / den[(size_t)dN * NHEAD + h];
    const float* vp = KQV + (size_t)sN * 192 + 128 + h * DHEAD;
    const float* vr = vrel + h * 64;
    float vv[8];
    #pragma unroll
    for (int i = 0; i < 8; ++i) vv[i] = vp[i];
    #pragma unroll
    for (int eo = 0; eo < 8; ++eo) {
        float ve = 0.0f;
        #pragma unroll
        for (int di = 0; di < 8; ++di) ve += vv[di] * vr[di * 8 + eo];
        atomicAdd(agg + (size_t)dN * HDIM + h * DHEAD + eo, ve * w);
    }
}

// ---------------------------------------------------------------------------
__global__ void k_col_mean(const float* __restrict__ X, int N, float* __restrict__ out) {
    __shared__ float sm[256];
    int col = blockIdx.x;                          // 64 blocks, one per column
    float s = 0.0f;
    for (int r = threadIdx.x; r < N; r += 256) s += X[(size_t)r * HDIM + col];
    sm[threadIdx.x] = s;
    __syncthreads();
    for (int st = 128; st > 0; st >>= 1) {
        if ((int)threadIdx.x < st) sm[threadIdx.x] += sm[threadIdx.x + st];
        __syncthreads();
    }
    if (threadIdx.x == 0) out[col] = sm[0] / (float)N;
}

__global__ void k_value_head(const float* __restrict__ g, const float* __restrict__ Wv1,
                             const float* __restrict__ bv1, const float* __restrict__ Wv2,
                             const float* __restrict__ bv2, float* __restrict__ out) {
    __shared__ float h1[64];
    int j = threadIdx.x;                           // 64 threads
    float a = bv1[j];
    for (int i = 0; i < 192; ++i) a += g[i] * Wv1[i * HDIM + j];
    h1[j] = fmaxf(a, 0.0f);
    __syncthreads();
    if (j == 0) {
        float v = bv2[0];
        for (int i = 0; i < 64; ++i) v += h1[i] * Wv2[i];
        out[0] = v;
    }
}

// policy-pair feature assembly, writes the GEMM A operand directly in f16
__global__ void k_gather_z(const float* __restrict__ X, const int* __restrict__ opIdx,
                           const int* __restrict__ mIdx, const float* __restrict__ g,
                           int P, int machOff, _Float16* __restrict__ Z) {
    long t = (long)blockIdx.x * blockDim.x + threadIdx.x;
    if (t >= (long)P * 320) return;
    int p = (int)(t / 320), j = (int)(t % 320);
    float v;
    if (j < 64)       v = X[(size_t)opIdx[p] * HDIM + j];
    else if (j < 128) v = X[(size_t)(machOff + mIdx[p]) * HDIM + (j - 64)];
    else              v = g[j - 128];
    Z[t] = (_Float16)v;
}

__global__ void k_logits(const float* __restrict__ Z2, const float* __restrict__ Wp3,
                         const float* __restrict__ bp3, float* __restrict__ out, int P) {
    int p = blockIdx.x * blockDim.x + threadIdx.x;
    if (p >= P) return;
    float v = bp3[0];
    #pragma unroll
    for (int j = 0; j < 32; ++j) v += Z2[(size_t)p * 32 + j] * Wp3[j];
    out[p] = v;
}

// ---------------------------------------------------------------------------
extern "C" void kernel_launch(void* const* d_in, const int* in_sizes, int n_in,
                              void* d_out, int out_size, void* d_ws, size_t ws_size,
                              hipStream_t stream) {
    const float* op_x   = (const float*)d_in[0];
    const float* mach_x = (const float*)d_in[1];
    const float* job_x  = (const float*)d_in[2];
    const float* W_op   = (const float*)d_in[3];
    const float* b_op   = (const float*)d_in[4];
    const float* W_mach = (const float*)d_in[5];
    const float* b_mach = (const float*)d_in[6];
    const float* W_job  = (const float*)d_in[7];
    const float* b_job  = (const float*)d_in[8];
    const float* ln_gamma = (const float*)d_in[9];
    const float* ln_beta  = (const float*)d_in[10];
    const float* Wkqv = (const float*)d_in[11];
    const float* bkqv = (const float*)d_in[12];
    const float* krel = (const float*)d_in[13];
    const float* vrel = (const float*)d_in[14];
    const float* prel = (const float*)d_in[15];
    const float* Wout = (const float*)d_in[16];
    const float* bout = (const float*)d_in[17];
    const float* skip = (const float*)d_in[18];
    const float* Wp1 = (const float*)d_in[19];
    const float* bp1 = (const float*)d_in[20];
    const float* Wp2 = (const float*)d_in[21];
    const float* bp2 = (const float*)d_in[22];
    const float* Wp3 = (const float*)d_in[23];
    const float* bp3 = (const float*)d_in[24];
    const float* Wv1 = (const float*)d_in[25];
    const float* bv1 = (const float*)d_in[26];
    const float* Wv2 = (const float*)d_in[27];
    const float* bv2 = (const float*)d_in[28];
    const int* edges[8];
    int Ecnt[8];
    long Etot = 0;
    for (int e = 0; e < 8; ++e) {
        edges[e] = (const int*)d_in[29 + e];
        Ecnt[e]  = in_sizes[29 + e] / 2;
        Etot += Ecnt[e];
    }
    const int* op_idx = (const int*)d_in[37];
    const int* m_idx  = (const int*)d_in[38];
    float* dout = (float*)d_out;

    const int OPF = in_sizes[3] / HDIM;
    const int MF  = in_sizes[5] / HDIM;
    const int JF  = in_sizes[7] / HDIM;
    const int NOP = in_sizes[0] / OPF;
    const int NM  = in_sizes[1] / MF;
    const int NJ  = in_sizes[2] / JF;
    const int L   = in_sizes[18] / 3;
    const int P   = in_sizes[37];
    const int Ntot = NOP + NM + NJ;
    const int off[3] = {0, NOP, NOP + NM};
    const int Ns[3]  = {NOP, NM, NJ};
    // EDGE_DEFS = ((2,0),(0,2),(0,0),(0,0),(0,1),(0,1),(1,0),(1,0))
    const int sOffT[8] = {off[2], off[0], off[0], off[0], off[0], off[0], off[1], off[1]};
    const int dOffT[8] = {off[0], off[2], off[0], off[0], off[1], off[1], off[0], off[0]};

    // ---- workspace carve-up: f32 regions then 16B-aligned f16 regions ----
    float* ws = (float*)d_ws;
    size_t o = 0;
    float* XA    = ws + o; o += (size_t)Ntot * HDIM;
    float* XB    = ws + o; o += (size_t)Ntot * HDIM;
    float* KQV   = ws + o; o += (size_t)Ntot * 3 * HDIM;
    float* ALPHA = ws + o; o += (size_t)Etot * NHEAD;
    float* AMAX  = ws + o; o += (size_t)Ntot * NHEAD;
    float* DEN   = ws + o; o += (size_t)Ntot * NHEAD;
    float* AGG   = ws + o; o += (size_t)Ntot * HDIM;
    float* SM    = ws + o; o += 256;                 // stats(6) + pooled g(192)
    o = (o + 15) & ~(size_t)15;
    _Float16* hb = (_Float16*)(ws + o);
    size_t ho = 0;
    _Float16* XH    = hb + ho; ho += (size_t)Ntot * HDIM;       // f16 activations
    _Float16* AGGH  = hb + ho; ho += (size_t)Ntot * HDIM;       // f16 gelu(agg)
    _Float16* Z1H   = hb + ho; ho += (size_t)P * HDIM;
    _Float16* WkqvH = hb + ho; ho += (size_t)L * 3 * 192 * HDIM; // [N=192][K=64] per (l,t)
    _Float16* WoutH = hb + ho; ho += (size_t)L * 3 * HDIM * HDIM;
    _Float16* Wp1H  = hb + ho; ho += (size_t)HDIM * 320;
    _Float16* Wp2H  = hb + ho; ho += (size_t)32 * HDIM;
    // post-layer aliases
    _Float16* ZH = XH;                               // [P,320] <= XH+AGGH region
    float* Z1 = ALPHA;                               // [P,64] f32
    float* Z2 = ALPHA + (size_t)P * HDIM;            // [P,32] f32
    float* G  = SM + 8;                              // [192]

    auto cdiv = [](long a, long b) { return (int)((a + b - 1) / b); };

    // ---- one-time (per call) weight prep: f16 transposed copies ----
    for (int lt = 0; lt < L * 3; ++lt) {
        k_wt16<<<cdiv((long)192 * HDIM, 256), 256, 0, stream>>>(
            Wkqv + (size_t)lt * HDIM * 192, WkqvH + (size_t)lt * 192 * HDIM, HDIM, 192);
        k_wt16<<<cdiv((long)HDIM * HDIM, 256), 256, 0, stream>>>(
            Wout + (size_t)lt * HDIM * HDIM, WoutH + (size_t)lt * HDIM * HDIM, HDIM, HDIM);
    }
    k_wt16<<<cdiv((long)HDIM * 320, 256), 256, 0, stream>>>(Wp1, Wp1H, 320, HDIM);
    k_wt16<<<cdiv((long)32 * HDIM, 256), 256, 0, stream>>>(Wp2, Wp2H, HDIM, 32);

    // ---- input projection + graph norm ----
    k_fill<<<1, 256, 0, stream>>>(SM, 8, 0.0f);
    k_proj_stats<<<cdiv((long)NOP * HDIM, 256), 256, 0, stream>>>(
        op_x, W_op, b_op, NOP, OPF, XA + (size_t)off[0] * HDIM, SM + 0);
    k_proj_stats<<<cdiv((long)NM * HDIM, 256), 256, 0, stream>>>(
        mach_x, W_mach, b_mach, NM, MF, XA + (size_t)off[1] * HDIM, SM + 2);
    k_proj_stats<<<cdiv((long)NJ * HDIM, 256), 256, 0, stream>>>(
        job_x, W_job, b_job, NJ, JF, XA + (size_t)off[2] * HDIM, SM + 4);
    for (int t = 0; t < 3; ++t)
        k_norm<<<cdiv((long)Ns[t] * HDIM, 256), 256, 0, stream>>>(
            XA + (size_t)off[t] * HDIM, Ns[t], SM + 2 * t,
            ln_gamma + t * HDIM, ln_beta + t * HDIM);

    // ---- HGT layers ----
    float* cur = XA;
    float* nxt = XB;
    for (int l = 0; l < L; ++l) {
        // stage activations in f16 for the WMMA GEMMs
        k_cvt16<<<cdiv((long)Ntot * HDIM, 1024), 256, 0, stream>>>(
            cur, XH, (long)Ntot * HDIM, 0);
        // KQV projection per type: [N_t,64] @ [64,192]  (WMMA, 16x32 per wave)
        for (int t = 0; t < 3; ++t) {
            int tiles = cdiv(Ns[t], 16) * (192 / 32);
            k_gemm_wmma<0><<<cdiv(tiles, 8), 256, 0, stream>>>(
                XH + (size_t)off[t] * HDIM,
                WkqvH + (size_t)(l * 3 + t) * 192 * HDIM,
                bkqv + (size_t)(l * 3 + t) * 192,
                KQV + (size_t)off[t] * 192,
                Ns[t], 192, HDIM, nullptr, nullptr);
        }
        // reset per-destination softmax state + aggregate buffer
        k_fill<<<cdiv((long)Ntot * NHEAD, 256), 256, 0, stream>>>(AMAX, (long)Ntot * NHEAD, -3.0e38f);
        k_fill<<<cdiv((long)Ntot * NHEAD, 256), 256, 0, stream>>>(DEN, (long)Ntot * NHEAD, 0.0f);
        k_fill<<<cdiv((long)Ntot * HDIM, 256), 256, 0, stream>>>(AGG, (long)Ntot * HDIM, 0.0f);
        // pass 1: alpha + segment max
        long aoff = 0;
        for (int e = 0; e < 8; ++e) {
            k_edge_alpha<<<cdiv((long)Ecnt[e] * NHEAD, 256), 256, 0, stream>>>(
                edges[e], Ecnt[e], sOffT[e], dOffT[e], KQV,
                krel + (size_t)(l * 8 + e) * 512, prel + (size_t)(l * 8 + e) * NHEAD,
                ALPHA + aoff * NHEAD, AMAX);
            aoff += Ecnt[e];
        }
        // pass 2: softmax denominator
        aoff = 0;
        for (int e = 0; e < 8; ++e) {
            k_edge_den<<<cdiv((long)Ecnt[e] * NHEAD, 256), 256, 0, stream>>>(
                edges[e], Ecnt[e], dOffT[e], ALPHA + aoff * NHEAD, AMAX, DEN);
            aoff += Ecnt[e];
        }
        // pass 3: weighted message aggregation
        aoff = 0;
        for (int e = 0; e < 8; ++e) {
            k_edge_agg<<<cdiv((long)Ecnt[e] * NHEAD, 256), 256, 0, stream>>>(
                edges[e], Ecnt[e], sOffT[e], dOffT[e], KQV,
                vrel + (size_t)(l * 8 + e) * 512,
                ALPHA + aoff * NHEAD, AMAX, DEN, AGG);
            aoff += Ecnt[e];
        }
        // gelu fused into the f16 staging pass for the out-projection A operand
        k_cvt16<<<cdiv((long)Ntot * HDIM, 1024), 256, 0, stream>>>(
            AGG, AGGH, (long)Ntot * HDIM, 1);
        // output projection + sigmoid(skip)-gated residual (WMMA)
        for (int t = 0; t < 3; ++t) {
            int tiles = cdiv(Ns[t], 16) * (HDIM / 32);
            k_gemm_wmma<2><<<cdiv(tiles, 8), 256, 0, stream>>>(
                AGGH + (size_t)off[t] * HDIM,
                WoutH + (size_t)(l * 3 + t) * HDIM * HDIM,
                bout + (size_t)(l * 3 + t) * HDIM,
                nxt + (size_t)off[t] * HDIM,
                Ns[t], HDIM, HDIM,
                cur + (size_t)off[t] * HDIM, skip + (size_t)l * 3 + t);
        }
        float* tmp = cur; cur = nxt; nxt = tmp;
    }

    // ---- global pooling + value head ----
    k_col_mean<<<64, 256, 0, stream>>>(cur + (size_t)off[0] * HDIM, NOP, G + 0);
    k_col_mean<<<64, 256, 0, stream>>>(cur + (size_t)off[1] * HDIM, NM, G + 64);
    k_col_mean<<<64, 256, 0, stream>>>(cur + (size_t)off[2] * HDIM, NJ, G + 128);
    k_value_head<<<1, 64, 0, stream>>>(G, Wv1, bv1, Wv2, bv2, dout + P);

    // ---- policy head: gather [op | machine | g] (f16) then WMMA MLP ----
    k_gather_z<<<cdiv((long)P * 320, 256), 256, 0, stream>>>(cur, op_idx, m_idx, G, P, NOP, ZH);
    {
        int tiles = cdiv(P, 16) * (HDIM / 32);
        k_gemm_wmma<1><<<cdiv(tiles, 8), 256, 0, stream>>>(
            ZH, Wp1H, bp1, Z1, P, HDIM, 320, nullptr, nullptr);
    }
    k_cvt16<<<cdiv((long)P * HDIM, 1024), 256, 0, stream>>>(Z1, Z1H, (long)P * HDIM, 0);
    {
        int tiles = cdiv(P, 16) * (32 / 32);
        k_gemm_wmma<1><<<cdiv(tiles, 8), 256, 0, stream>>>(
            Z1H, Wp2H, bp2, Z2, P, 32, HDIM, nullptr, nullptr);
    }
    k_logits<<<cdiv(P, 256), 256, 0, stream>>>(Z2, Wp3, bp3, dout, P);
}
